// GraphTransformerNet_29257317220697
// MI455X (gfx1250) — compile-verified, matrix operand order checked by reading, not run on hardware
//
#include <hip/hip_runtime.h>

// ---------------------------------------------------------------------------
// GraphTransformerNet for gfx1250 (MI455X), wave32 + WMMA f16->f32.
// - Dense GEMMs: v_wmma_f32_16x16x32_f16, one 16x16 C tile per wave,
//   8 waves (128 rows) per block, A fragments hoisted in VGPRs across all
//   column tiles. All operand loads are unconditional global_load_b128 from
//   zero-padded f16 staging buffers (EXEC all-ones around WMMA).
// - QKV fused into one GEMM against the contiguous [240][96] packed weight.
// - Epilogue variants (bias/residual/relu/f32/f16/colmask) are compile-time
//   template parameters; full tiles take a wave-uniform fast path with
//   unconditional stores (no EXEC churn), only the tail block masks rows.
// - Edge-restricted softmax aggregation via fp32 global atomics (L2-resident).
// ---------------------------------------------------------------------------

typedef __attribute__((ext_vector_type(16))) _Float16 v16h;
typedef __attribute__((ext_vector_type(8)))  _Float16 v8h;
typedef __attribute__((ext_vector_type(8)))  float    v8f;

#define DHID 80
#define NHEAD 8
#define DKH 10

static __host__ __device__ inline int ceil_div_i(int a, int b) { return (a + b - 1) / b; }

// -------------------------- small elementwise kernels ----------------------

__global__ void embed_kernel(const float* __restrict__ x,
                             const float* __restrict__ We,
                             const float* __restrict__ be,
                             float* __restrict__ h, int N) {
  int i = blockIdx.x * blockDim.x + threadIdx.x;
  if (i >= N * DHID) return;
  int row = i / DHID, c = i - row * DHID;
  float acc = be[c];
  #pragma unroll
  for (int k = 0; k < 9; ++k) acc += x[row * 9 + k] * We[k * DHID + c];
  h[i] = acc;
}

// fp32 [N,Kin] -> f16 zero-padded [Mpad,KPAD]; 8 halfs (one b128) per thread
__global__ void cvt_pad8_kernel(const float* __restrict__ in, _Float16* __restrict__ out,
                                int N, int Kin, int Mpad, int KPAD) {
  int i = blockIdx.x * blockDim.x + threadIdx.x;
  int k8 = KPAD >> 3;
  int total = Mpad * k8;
  if (i >= total) return;
  int row = i / k8;
  int k0 = (i - row * k8) << 3;
  v8h o;
  #pragma unroll
  for (int t = 0; t < 8; ++t) {
    int k = k0 + t;
    float v = (row < N && k < Kin) ? in[(size_t)row * Kin + k] : 0.0f;
    o[t] = (_Float16)v;
  }
  *(v8h*)(out + (size_t)row * KPAD + k0) = o;
}

// h_attn = wV/(z+1e-6) -> f16 padded [Mpad,96]; 8 halfs per thread
__global__ void norm_cvt8_kernel(const float* __restrict__ wV, const float* __restrict__ z,
                                 _Float16* __restrict__ out, int N, int Mpad) {
  int i = blockIdx.x * blockDim.x + threadIdx.x;
  int total = Mpad * 12;                 // 96/8
  if (i >= total) return;
  int row = i / 12;
  int k0 = (i - row * 12) << 3;
  v8h o;
  #pragma unroll
  for (int t = 0; t < 8; ++t) {
    int k = k0 + t;
    float v = 0.0f;
    if (row < N && k < DHID) {
      int hh = k / DKH;
      v = wV[(size_t)row * DHID + k] / (z[(size_t)row * NHEAD + hh] + 1e-6f);
    }
    o[t] = (_Float16)v;
  }
  *(v8h*)(out + (size_t)row * 96 + k0) = o;
}

__global__ void fill_f32_kernel(float* __restrict__ p, float v, int n) {
  int i = blockIdx.x * blockDim.x + threadIdx.x;
  if (i < n) p[i] = v;
}

// zero the f16 padding columns [160,192) of a [Mpad,192] buffer (b128 stores)
__global__ void zero_pad192_kernel(_Float16* __restrict__ buf, int Mpad) {
  int i = blockIdx.x * blockDim.x + threadIdx.x;
  int total = Mpad * 4;
  if (i >= total) return;
  int row = i >> 2, c = (i & 3) << 3;
  v8h zv = {};
  *(v8h*)(buf + (size_t)row * 192 + 160 + c) = zv;
}

// W fp32 [K,Nc] row-major -> WT f16 [Ncpad,KPAD] (transposed, zero padded)
__global__ void prep_wt_kernel(const float* __restrict__ W, _Float16* __restrict__ WT,
                               int K, int Nc, int Ncpad, int KPAD) {
  int i = blockIdx.x * blockDim.x + threadIdx.x;
  int total = Ncpad * KPAD;
  if (i >= total) return;
  int n = i / KPAD, k = i - n * KPAD;
  float v = (n < Nc && k < K) ? W[(size_t)k * Nc + n] : 0.0f;
  WT[i] = (_Float16)v;
}

// ------------------------------ edge attention -----------------------------
// qkv layout per node: [Q(80) | K(80) | V(80)] stride 240.

__global__ void attn_edge_kernel(const float* __restrict__ qkv,
                                 const int* __restrict__ src, const int* __restrict__ dst,
                                 float* __restrict__ wV, float* __restrict__ z, int E) {
  int i = blockIdx.x * blockDim.x + threadIdx.x;
  if (i >= E * NHEAD) return;
  int e = i >> 3, hh = i & 7;
  int s = src[e], d = dst[e];
  const float* qp = qkv + (size_t)d * 240 + hh * DKH;          // Q[dst]
  const float* kp = qkv + (size_t)s * 240 + 80 + hh * DKH;     // K[src]
  const float* vp = qkv + (size_t)s * 240 + 160 + hh * DKH;    // V[src]
  float dot = 0.0f;
  #pragma unroll
  for (int t = 0; t < DKH; ++t) dot += kp[t] * qp[t];
  dot *= 0.31622776601683794f;            // 1/sqrt(10)
  dot = fminf(5.0f, fmaxf(-5.0f, dot));   // clamp then exp (matches reference)
  float sc = expf(dot);
  float* wp = wV + (size_t)d * DHID + hh * DKH;
  #pragma unroll
  for (int t = 0; t < DKH; ++t) atomicAdd(&wp[t], vp[t] * sc);
  atomicAdd(&z[(size_t)d * NHEAD + hh], sc);
}

// ------------------------------- WMMA GEMM ---------------------------------
// C[N,Nc] = A[Mpad,KPAD] (f16, padded) x BT[Ncpad,KPAD] (f16, transposed).
// Epilogue fully selected at compile time.
//
// A fragment layout (16-bit A 16x32, ISA 7.12.2): lane<16 holds row M=lane,
// K=base..base+7 then base+16..base+23 with base = kk + 8*(lane>>4); B is
// loaded from the pre-transposed weight with the mirrored layout.
// C layout: element (M = r + 8*(lane>>4), Ncol = lane&15) in VGPR r.

template <int NCHUNKS, bool BIAS, bool RESID, bool RELU, bool OUTF32, bool OUTF16, bool COLMASK>
__global__ __launch_bounds__(256) void gemm_wmma_kernel(
    const _Float16* __restrict__ A, const _Float16* __restrict__ BT,
    const float* __restrict__ bias, const float* __restrict__ resid,
    float* __restrict__ Cf32, _Float16* __restrict__ Cf16,
    int N, int Nc, int ldc16) {
  constexpr int KPAD = 32 * NCHUNKS;
  const int lane  = threadIdx.x & 31;
  const int wave  = threadIdx.x >> 5;
  const int row0  = blockIdx.x * 128 + wave * 16;
  const int m     = lane & 15;
  const int khalf = lane >> 4;

  // Hoist A fragments for this wave's 16 rows across all K chunks.
  v16h afrag[NCHUNKS];
  const _Float16* arow = A + (size_t)(row0 + m) * KPAD + khalf * 8;
  #pragma unroll
  for (int c = 0; c < NCHUNKS; ++c) {
    union { v16h v; v8h h[2]; } u;
    u.h[0] = *(const v8h*)(arow + c * 32);
    u.h[1] = *(const v8h*)(arow + c * 32 + 16);
    afrag[c] = u.v;
  }

  const int ncoltiles = (Nc + 15) >> 4;
  for (int ct = 0; ct < ncoltiles; ++ct) {
    const _Float16* brow = BT + (size_t)(ct * 16 + m) * KPAD + khalf * 8;
    v8f acc = {};
    #pragma unroll
    for (int c = 0; c < NCHUNKS; ++c) {
      union { v16h v; v8h h[2]; } u;
      u.h[0] = *(const v8h*)(brow + c * 32);
      u.h[1] = *(const v8h*)(brow + c * 32 + 16);
      acc = __builtin_amdgcn_wmma_f32_16x16x32_f16(
          /*neg_a=*/false, afrag[c], /*neg_b=*/false, u.v,
          /*c_mod=*/(short)0, acc, /*reuse_a=*/false, /*reuse_b=*/false);
    }
    const int col = ct * 16 + m;
    const bool colok = COLMASK ? (col < Nc) : true;
    float bv = 0.0f;
    if (BIAS) bv = bias[COLMASK ? (colok ? col : 0) : col];

    if (!COLMASK && (row0 + 16 <= N)) {
      // Full tile: wave-uniform fast path, unconditional stores.
      #pragma unroll
      for (int r = 0; r < 8; ++r) {
        const int row = row0 + r + 8 * khalf;
        float v = acc[r];
        if (BIAS)  v += bv;
        if (RESID) v += resid[(size_t)row * Nc + col];
        if (RELU)  v = fmaxf(v, 0.0f);
        if (OUTF32) Cf32[(size_t)row * Nc + col] = v;
        if (OUTF16) Cf16[(size_t)row * ldc16 + col] = (_Float16)v;
      }
    } else {
      // Tail block (or ragged Nc): per-lane masked stores.
      #pragma unroll
      for (int r = 0; r < 8; ++r) {
        const int row = row0 + r + 8 * khalf;
        if ((row < N) && colok) {
          float v = acc[r];
          if (BIAS)  v += bv;
          if (RESID) v += resid[(size_t)row * Nc + col];
          if (RELU)  v = fmaxf(v, 0.0f);
          if (OUTF32) Cf32[(size_t)row * Nc + col] = v;
          if (OUTF16) Cf16[(size_t)row * ldc16 + col] = (_Float16)v;
        }
      }
    }
  }
}

// ----------------------------------------------------------------------------

extern "C" void kernel_launch(void* const* d_in, const int* in_sizes, int n_in,
                              void* d_out, int out_size, void* d_ws, size_t ws_size,
                              hipStream_t stream) {
  (void)n_in; (void)out_size; (void)ws_size;
  const float* x   = (const float*)d_in[0];
  const int*   ei  = (const int*)d_in[1];
  const float* We  = (const float*)d_in[2];
  const float* be  = (const float*)d_in[3];
  const float* Wq  = (const float*)d_in[4];
  const float* Wk  = (const float*)d_in[5];
  const float* Wv  = (const float*)d_in[6];
  const float* Wo  = (const float*)d_in[7];
  const float* bo  = (const float*)d_in[8];
  const float* W1  = (const float*)d_in[9];
  const float* b1  = (const float*)d_in[10];
  const float* W2  = (const float*)d_in[11];
  const float* b2  = (const float*)d_in[12];
  const float* Wr0 = (const float*)d_in[13];
  const float* br0 = (const float*)d_in[14];
  const float* Wr1 = (const float*)d_in[15];
  const float* br1 = (const float*)d_in[16];
  const float* Wr2 = (const float*)d_in[17];
  const float* br2 = (const float*)d_in[18];
  float* out = (float*)d_out;

  const int N = in_sizes[0] / 9;
  const int E = in_sizes[1] / 2;
  const int L = in_sizes[4] / (DHID * DHID);
  const int Mpad = ceil_div_i(N, 128) * 128;
  const int nblocks = Mpad / 128;
  const int* srcIdx = ei;       // edge_index[0]
  const int* dstIdx = ei + E;   // edge_index[1]

  // --------------------------- workspace carving ---------------------------
  char* cur = (char*)d_ws;
  auto wsAlloc = [&](size_t bytes) -> void* {
    void* r = (void*)cur;
    cur += (bytes + 255) & ~(size_t)255;
    return r;
  };
  float*     h    = (float*)wsAlloc((size_t)N * DHID * 4);
  _Float16*  bufA = (_Float16*)wsAlloc((size_t)Mpad * 192 * 2);
  _Float16*  bufB = (_Float16*)wsAlloc((size_t)Mpad * 192 * 2);
  float*     qkv  = (float*)wsAlloc((size_t)N * 240 * 4);
  float*     wV   = (float*)wsAlloc((size_t)N * DHID * 4);
  float*     zacc = (float*)wsAlloc((size_t)N * NHEAD * 4);
  float*     r0   = (float*)wsAlloc((size_t)N * 40 * 4);
  float*     r1   = (float*)wsAlloc((size_t)N * 20 * 4);
  // transposed+padded f16 weights, per layer:
  //   [WqT|WkT|WvT] contiguous = fused QKV weight [240][96], WoT [80][96],
  //   W1T [160][96], W2T [80][192]
  const size_t W80_96  = (size_t)80 * 96;
  const size_t W160_96 = (size_t)160 * 96;
  const size_t W80_192 = (size_t)80 * 192;
  const size_t LAYER_H = 4 * W80_96 + W160_96 + W80_192;
  _Float16* wts   = (_Float16*)wsAlloc((size_t)L * LAYER_H * 2);
  _Float16* Wr0T  = (_Float16*)wsAlloc((size_t)48 * 96 * 2);
  _Float16* Wr1T  = (_Float16*)wsAlloc((size_t)32 * 64 * 2);
  _Float16* Wr2T  = (_Float16*)wsAlloc((size_t)16 * 32 * 2);

  const int TPB = 256;
  auto grid1 = [&](int n) { return dim3((unsigned)ceil_div_i(n, TPB)); };
  dim3 gGemm((unsigned)nblocks), bGemm(256);

  // ---------------------------- weight preparation -------------------------
  for (int l = 0; l < L; ++l) {
    _Float16* base = wts + (size_t)l * LAYER_H;
    prep_wt_kernel<<<grid1(80 * 96), TPB, 0, stream>>>(Wq + (size_t)l * 6400, base,              80, 80, 80, 96);
    prep_wt_kernel<<<grid1(80 * 96), TPB, 0, stream>>>(Wk + (size_t)l * 6400, base + W80_96,     80, 80, 80, 96);
    prep_wt_kernel<<<grid1(80 * 96), TPB, 0, stream>>>(Wv + (size_t)l * 6400, base + 2 * W80_96, 80, 80, 80, 96);
    prep_wt_kernel<<<grid1(80 * 96), TPB, 0, stream>>>(Wo + (size_t)l * 6400, base + 3 * W80_96, 80, 80, 80, 96);
    prep_wt_kernel<<<grid1(160 * 96), TPB, 0, stream>>>(W1 + (size_t)l * 12800, base + 4 * W80_96, 80, 160, 160, 96);
    prep_wt_kernel<<<grid1(80 * 192), TPB, 0, stream>>>(W2 + (size_t)l * 12800, base + 4 * W80_96 + W160_96, 160, 80, 80, 192);
  }
  prep_wt_kernel<<<grid1(48 * 96), TPB, 0, stream>>>(Wr0, Wr0T, 80, 40, 48, 96);
  prep_wt_kernel<<<grid1(32 * 64), TPB, 0, stream>>>(Wr1, Wr1T, 40, 20, 32, 64);
  prep_wt_kernel<<<grid1(16 * 32), TPB, 0, stream>>>(Wr2, Wr2T, 20, 4, 16, 32);

  // ------------------------------- embedding -------------------------------
  embed_kernel<<<grid1(N * DHID), TPB, 0, stream>>>(x, We, be, h, N);

  // -------------------------------- layers ---------------------------------
  for (int l = 0; l < L; ++l) {
    _Float16* WqkvT = wts + (size_t)l * LAYER_H;          // [240][96]
    _Float16* WoT   = WqkvT + 3 * W80_96;                 // [80][96]
    _Float16* W1T   = WqkvT + 4 * W80_96;                 // [160][96]
    _Float16* W2T   = W1T + W160_96;                      // [80][192]
    const float* bo_l = bo + (size_t)l * DHID;
    const float* b1_l = b1 + (size_t)l * 160;
    const float* b2_l = b2 + (size_t)l * DHID;

    // h (fp32) -> bufA (f16, [Mpad,96])
    cvt_pad8_kernel<<<grid1(Mpad * 12), TPB, 0, stream>>>(h, bufA, N, DHID, Mpad, 96);

    // fused Q|K|V projection: [N,80] x [80,240] -> qkv [N,240]
    gemm_wmma_kernel<3, false, false, false, true, false, false>
        <<<gGemm, bGemm, 0, stream>>>(bufA, WqkvT, nullptr, nullptr, qkv, nullptr, N, 240, 0);

    // edge-restricted softmax aggregation
    fill_f32_kernel<<<grid1(N * DHID), TPB, 0, stream>>>(wV, 0.0f, N * DHID);
    fill_f32_kernel<<<grid1(N * NHEAD), TPB, 0, stream>>>(zacc, 0.0f, N * NHEAD);
    attn_edge_kernel<<<grid1(E * NHEAD), TPB, 0, stream>>>(qkv, srcIdx, dstIdx, wV, zacc, E);

    // h_attn = wV/(z+1e-6) -> bufB (f16, [Mpad,96])
    norm_cvt8_kernel<<<grid1(Mpad * 12), TPB, 0, stream>>>(wV, zacc, bufB, N, Mpad);

    // O-projection, fused bias + residual, in-place on h
    gemm_wmma_kernel<3, true, true, false, true, false, false>
        <<<gGemm, bGemm, 0, stream>>>(bufB, WoT, bo_l, h, h, nullptr, N, DHID, 0);

    // FFN: h -> bufA f16; FFN1 writes ReLU f16 directly into bufB [Mpad,192]
    cvt_pad8_kernel<<<grid1(Mpad * 12), TPB, 0, stream>>>(h, bufA, N, DHID, Mpad, 96);
    zero_pad192_kernel<<<grid1(Mpad * 4), TPB, 0, stream>>>(bufB, Mpad);
    gemm_wmma_kernel<3, true, false, true, false, true, false>
        <<<gGemm, bGemm, 0, stream>>>(bufA, W1T, b1_l, nullptr, nullptr, bufB, N, 160, 192);
    // FFN2, fused bias + residual, in-place on h
    gemm_wmma_kernel<6, true, true, false, true, false, false>
        <<<gGemm, bGemm, 0, stream>>>(bufB, W2T, b2_l, h, h, nullptr, N, DHID, 0);
  }

  // ------------------------------- readout ---------------------------------
  cvt_pad8_kernel<<<grid1(Mpad * 12), TPB, 0, stream>>>(h, bufA, N, DHID, Mpad, 96);
  gemm_wmma_kernel<3, true, false, true, true, false, true>
      <<<gGemm, bGemm, 0, stream>>>(bufA, Wr0T, br0, nullptr, r0, nullptr, N, 40, 0);
  cvt_pad8_kernel<<<grid1(Mpad * 8), TPB, 0, stream>>>(r0, bufB, N, 40, Mpad, 64);
  gemm_wmma_kernel<2, true, false, true, true, false, true>
      <<<gGemm, bGemm, 0, stream>>>(bufB, Wr1T, br1, nullptr, r1, nullptr, N, 20, 0);
  cvt_pad8_kernel<<<grid1(Mpad * 4), TPB, 0, stream>>>(r1, bufA, N, 20, Mpad, 32);
  gemm_wmma_kernel<1, true, false, false, true, false, true>
      <<<gGemm, bGemm, 0, stream>>>(bufA, Wr2T, br2, nullptr, out, nullptr, N, 4, 0);
}
